// DeformConv_3882650436005
// MI455X (gfx1250) — matrix operand order repaired and checked by reading, not compiled, not run on hardware
//
#include <hip/hip_runtime.h>
#include <hip/hip_bf16.h>

typedef __attribute__((ext_vector_type(16))) _Float16 v16h;
typedef __attribute__((ext_vector_type(8)))  float    v8f;

// Problem constants (from reference setup_inputs)
#define BATCH 4
#define CIN   256
#define HH    64
#define WW    64
#define OUTC  256
#define KK2   9                 // 3x3 taps
#define KRED  (CIN * KK2)       // 2304 reduction length
#define NPIX  (HH * WW)         // 4096 pixels per batch

// Tiling
#define BM 128                  // output-channel tile
#define BN 128                  // pixel tile
#define BK 32                   // reduction chunk (one wmma K)
#define LDSROW 40               // padded row stride in halves (80B: 16B-aligned, bank-conflict-free)

union FragU { uint4 u4[2]; v16h h; };
union PackU { _Float16 h[2]; unsigned u; };

// ---- Phase 0: repack weights f32 [O][C][9] -> f16 [O][kk*256+c] (row-major, K-contiguous) ----
__global__ __launch_bounds__(256)
void repack_weights(const float* __restrict__ w, _Float16* __restrict__ wh) {
    const int t = blockIdx.x * 256 + threadIdx.x;
    if (t >= OUTC * KRED) return;
    const int o  = t / KRED;
    const int r  = t - o * KRED;     // kk*256 + c
    const int kk = r >> 8;
    const int c  = r & 255;
    wh[(size_t)o * KRED + r] = (_Float16)w[(size_t)o * KRED + (size_t)c * KK2 + kk];
}

// ---- Phase 1: fused bilinear-im2col + WMMA GEMM ----
__global__ __launch_bounds__(256)
void deform_conv_wmma(const float* __restrict__ x,
                      const float* __restrict__ offset,
                      const _Float16* __restrict__ wh,
                      float* __restrict__ out) {
    __shared__ _Float16 B_lds[2][BN * LDSROW];    // 2 x 10 KB ping-pong sampled-cols tile [p][k]
    __shared__ int   s_idx[KK2][4][BN];           // 18 KB clamped flat index per corner
    __shared__ float s_w  [KK2][4][BN];           // 18 KB masked bilinear weight per corner

    const int tid  = threadIdx.x;
    const int lane = tid & 31;
    const int wv   = tid >> 5;          // 0..7
    const int waveM = wv >> 2;          // 0..1  -> 64 rows each
    const int waveN = wv & 3;           // 0..3  -> 32 cols each
    const int lrow  = lane & 15;

    const int blk  = blockIdx.x;        // 0..255
    const int b    = blk >> 6;          // batch
    const int rem  = blk & 63;
    const int o0   = (rem >> 5) * BM;   // 0 or 128
    const int n0   = (rem & 31) * BN;   // pixel tile base

    // ---- Precompute bilinear sampling data for all 9 taps x 128 pixels ----
    for (int e = tid; e < KK2 * BN; e += 256) {
        const int kk = e >> 7;          // tap 0..8
        const int p  = e & 127;         // local pixel
        const int pg = n0 + p;          // global pixel
        const int ho = pg >> 6;
        const int wo = pg & 63;
        const float offy = offset[((size_t)(b * 18 + kk * 2 + 0)) * NPIX + pg];
        const float offx = offset[((size_t)(b * 18 + kk * 2 + 1)) * NPIX + pg];
        const float py = (float)(ho - 1 + kk / 3) + offy;
        const float px = (float)(wo - 1 + kk % 3) + offx;
        const float y0f = floorf(py), x0f = floorf(px);
        const float dy = py - y0f, dx = px - x0f;
        const int y0 = (int)y0f, x0 = (int)x0f;

        const int   yi[4] = { y0, y0, y0 + 1, y0 + 1 };
        const int   xi[4] = { x0, x0 + 1, x0, x0 + 1 };
        const float wc[4] = { (1.f - dy) * (1.f - dx), (1.f - dy) * dx,
                              dy * (1.f - dx),         dy * dx };
        #pragma unroll
        for (int c = 0; c < 4; ++c) {
            const bool valid = (yi[c] >= 0) & (yi[c] < HH) & (xi[c] >= 0) & (xi[c] < WW);
            int yc = yi[c] < 0 ? 0 : (yi[c] > HH - 1 ? HH - 1 : yi[c]);
            int xc = xi[c] < 0 ? 0 : (xi[c] > WW - 1 ? WW - 1 : xi[c]);
            s_idx[kk][c][p] = yc * WW + xc;          // always in-bounds (clamped)
            s_w  [kk][c][p] = valid ? wc[c] : 0.f;   // mask via weight
        }
    }
    __syncthreads();

    v8f acc[4][2];
    #pragma unroll
    for (int mi = 0; mi < 4; ++mi)
        #pragma unroll
        for (int ni = 0; ni < 2; ++ni)
            acc[mi][ni] = (v8f){0.f, 0.f, 0.f, 0.f, 0.f, 0.f, 0.f, 0.f};

    const int kbA = (lane < 16) ? 0 : 8;    // A: K at {kbA..kbA+7, kbA+16..kbA+23}
    const int kbB = (lane < 16) ? 0 : 8;    // B: K at {kbB..kbB+7, kbB+16..kbB+23}
    const int myp = tid & 127;              // this thread's pixel (invariant across chunks)
    const int cl0 = (tid >> 7) << 1;        // channel pair base: 0 or 2

    // Global row bases for this lane's 4 A-fragment rows (f16 [O][KRED], K-contiguous)
    const _Float16* aRow[4];
    #pragma unroll
    for (int mi = 0; mi < 4; ++mi)
        aRow[mi] = wh + (size_t)(o0 + waveM * 64 + mi * 16 + lrow) * KRED;

    for (int kk = 0; kk < KK2; ++kk) {
        // Cache this tap's sampling data for my pixel in registers (8 VGPRs)
        int   sidx[4];
        float sw[4];
        #pragma unroll
        for (int cn = 0; cn < 4; ++cn) {
            sidx[cn] = s_idx[kk][cn][myp];
            sw[cn]   = s_w[kk][cn][myp];
        }

        #pragma unroll 2
        for (int cc = 0; cc < 8; ++cc) {
            const int chunk = kk * 8 + cc;
            const int c0 = cc << 5;            // channel base within tap
            const int r0 = chunk << 5;         // reduction base
            _Float16* Bbuf = B_lds[chunk & 1];

            // ---- Fill B tile (ping-pong): bilinear-sample x on the fly -> f16 [p][k] ----
            #pragma unroll
            for (int i = 0; i < 8; ++i) {
                const int cl = cl0 + (i << 2);             // pairs (cl, cl+1)
                const float* xc0 = x + ((size_t)(b * CIN + c0 + cl)) * NPIX;
                const float* xc1 = xc0 + NPIX;
                float v0 = 0.f, v1 = 0.f;
                #pragma unroll
                for (int cn = 0; cn < 4; ++cn) {
                    v0 += sw[cn] * xc0[sidx[cn]];
                    v1 += sw[cn] * xc1[sidx[cn]];
                }
                PackU pk;
                pk.h[0] = (_Float16)v0;
                pk.h[1] = (_Float16)v1;
                *reinterpret_cast<unsigned*>(&Bbuf[myp * LDSROW + cl]) = pk.u;
            }
            __syncthreads();   // fill(chunk) done; safe: prior reads of this buffer
                               // completed before each wave's previous barrier

            // ---- A fragments straight from global (two b128 each), B from LDS ----
            FragU afrag[4], bfrag[2];
            #pragma unroll
            for (int mi = 0; mi < 4; ++mi) {
                afrag[mi].u4[0] = *reinterpret_cast<const uint4*>(aRow[mi] + r0 + kbA);
                afrag[mi].u4[1] = *reinterpret_cast<const uint4*>(aRow[mi] + r0 + kbA + 16);
            }
            #pragma unroll
            for (int ni = 0; ni < 2; ++ni) {
                const int n = waveN * 32 + ni * 16 + lrow;
                bfrag[ni].u4[0] = *reinterpret_cast<const uint4*>(&Bbuf[n * LDSROW + kbB]);
                bfrag[ni].u4[1] = *reinterpret_cast<const uint4*>(&Bbuf[n * LDSROW + kbB + 16]);
            }
            #pragma unroll
            for (int mi = 0; mi < 4; ++mi)
                #pragma unroll
                for (int ni = 0; ni < 2; ++ni)
                    acc[mi][ni] = __builtin_amdgcn_wmma_f32_16x16x32_f16(
                        false, afrag[mi].h, false, bfrag[ni].h,
                        (short)0, acc[mi][ni], false, false);
        }
    }

    // ---- Write back: C/D layout -> lane holds col L%16, rows j + 8*(L>=16) ----
    const int rsel = (lane >> 4) << 3;   // 0 or 8
    #pragma unroll
    for (int mi = 0; mi < 4; ++mi) {
        const int mbase = o0 + waveM * 64 + mi * 16 + rsel;
        #pragma unroll
        for (int ni = 0; ni < 2; ++ni) {
            const int ncol = n0 + waveN * 32 + ni * 16 + lrow;
            #pragma unroll
            for (int j = 0; j < 8; ++j) {
                out[((size_t)(b * OUTC + mbase + j)) * NPIX + ncol] = acc[mi][ni][j];
            }
        }
    }
}

extern "C" void kernel_launch(void* const* d_in, const int* in_sizes, int n_in,
                              void* d_out, int out_size, void* d_ws, size_t ws_size,
                              hipStream_t stream) {
    (void)in_sizes; (void)n_in; (void)out_size; (void)ws_size;
    const float* x      = (const float*)d_in[0];
    const float* offset = (const float*)d_in[1];
    const float* weight = (const float*)d_in[2];
    float* out = (float*)d_out;
    _Float16* wh = (_Float16*)d_ws;     // needs OUTC*KRED*2 = 1.18 MB scratch

    // Phase 0: weight repack to WMMA-friendly f16 layout
    repack_weights<<<(OUTC * KRED + 255) / 256, 256, 0, stream>>>(weight, wh);

    // Phase 1: 4 batches x 2 M-tiles x 32 N-tiles = 256 workgroups, 256 threads (8 waves)
    deform_conv_wmma<<<BATCH * 2 * 32, 256, 0, stream>>>(x, offset, wh, out);
}
